// GAT_12214886989935
// MI455X (gfx1250) — matrix operand order repaired and checked by reading, not compiled, not run on hardware
//
#include <hip/hip_runtime.h>

// ---------------------------------------------------------------------------
// GAT (3-layer, 8 heads) for MI455X / gfx1250.
//  - GEMMs via v_wmma_f32_16x16x32_bf16 (bf16 in, f32 accumulate)
//  - segment softmax + aggregation via L2-resident f32 atomics
// ---------------------------------------------------------------------------

typedef __attribute__((ext_vector_type(16))) __bf16 v16bf;
typedef __attribute__((ext_vector_type(8)))  __bf16 v8bf;
typedef __attribute__((ext_vector_type(8)))  float  v8f;

#define NEG_SLOPE 0.2f

static __device__ __forceinline__ unsigned f2ord(float f) {
  unsigned u = __float_as_uint(f);
  return (u & 0x80000000u) ? ~u : (u | 0x80000000u);   // order-preserving
}
static __device__ __forceinline__ float ord2f(unsigned k) {
  return __uint_as_float((k & 0x80000000u) ? (k & 0x7FFFFFFFu) : ~k);
}

// ---------------- utility: zero 32-bit words ----------------
__global__ void k_zero32(unsigned* __restrict__ p, long long n) {
  long long i = blockIdx.x * 256LL + threadIdx.x;
  if (i < n) p[i] = 0u;
}

// ---------------- layer-1 input: (N x 12 f32) -> (N x 32 bf16, zero pad) ----
__global__ void k_cvt_x(const float* __restrict__ x, __bf16* __restrict__ xbf,
                        long long N) {
  long long i = blockIdx.x * 256LL + threadIdx.x;
  if (i >= N * 32) return;
  long long n = i >> 5;
  int k = (int)(i & 31);
  float v = (k < 12) ? x[n * 12 + k] : 0.0f;
  xbf[i] = (__bf16)v;
}

// ------- W (Kin x Nin f32 row-major) -> Wt (Nrows x Kpad bf16, transposed+pad)
__global__ void k_cvt_wt(const float* __restrict__ W, __bf16* __restrict__ Wt,
                         int Kin, int Nin, int Kpad, int Nrows) {
  long long i = blockIdx.x * 256LL + threadIdx.x;
  if (i >= (long long)Nrows * Kpad) return;
  int nrow = (int)(i / Kpad);
  int k    = (int)(i % Kpad);
  float v = (nrow < Nin && k < Kin) ? W[(size_t)k * Nin + nrow] : 0.0f;
  Wt[i] = (__bf16)v;
}

// ---------------- WMMA GEMM: C[MxN] = A[MxK] * BT[NxK]^T -------------------
// A row-major bf16 (lda = padded K), BT row-major bf16 over K (ldb = padded K).
// One wave per 16x16 output tile; K consumed 32 at a time.
//
// A fragment (ISA 7.12.2, 16-bit A 16x32): lane l16 = row; lane half owns
//   K in [khi, khi+8) and [16+khi, 16+khi+8), khi = 8*half.
// B fragment (ISA 7.12.2/7.12.4, 16-bit B 32x16): lane l16 = column; lane
//   half owns K in [kb, kb+8) and [16+kb, 16+kb+8), kb = 8*half (same as A).
// Both fragments -> two 16B global loads per lane per K-step.
__global__ void k_gemm_bf16(const __bf16* __restrict__ A,
                            const __bf16* __restrict__ BT,
                            float* __restrict__ C,
                            int Mtiles, int Ntiles, int Ksteps,
                            int lda, int ldb, int ldc, int nlim) {
  int wave = blockIdx.x * (int)(blockDim.x >> 5) + (int)(threadIdx.x >> 5);
  if (wave >= Mtiles * Ntiles) return;          // whole-wave exit: EXEC all-1s
  int mt = wave / Ntiles, nt = wave % Ntiles;
  int lane = threadIdx.x & 31;
  int half = lane >> 4;
  int l16  = lane & 15;
  int khi  = half * 8;    // A half-offset
  int kb   = half * 8;    // B half-offset (same as A)

  const __bf16* ap = A  + (size_t)(mt * 16 + l16) * lda + khi;
  const __bf16* bp = BT + (size_t)(nt * 16 + l16) * ldb + kb;

  v8f c = {};
  for (int ks = 0; ks < Ksteps; ++ks) {
    v8bf alo = *(const v8bf*)(ap);        // A: K = k0+khi    .. +7
    v8bf ahi = *(const v8bf*)(ap + 16);   // A: K = k0+16+khi .. +7
    v8bf blo = *(const v8bf*)(bp);        // B: K = k0+kb     .. +7
    v8bf bhi = *(const v8bf*)(bp + 16);   // B: K = k0+16+kb  .. +7
    v16bf a, b;
#pragma unroll
    for (int i = 0; i < 8; ++i) {
      a[i] = alo[i]; a[i + 8] = ahi[i];
      b[i] = blo[i]; b[i + 8] = bhi[i];
    }
    c = __builtin_amdgcn_wmma_f32_16x16x32_bf16(
            /*neg_a=*/false, a, /*neg_b=*/false, b,
            /*c_mod=*/(short)0, c, /*reuse_a=*/false, /*reuse_b=*/false);
    ap += 32; bp += 32;
  }

  // C/D layout: VGPR r holds M = mt*16 + r + 8*half, col = nt*16 + l16
  int col = nt * 16 + l16;
#pragma unroll
  for (int r = 0; r < 8; ++r) {
    int row = mt * 16 + r + 8 * half;
    if (col < nlim) C[(size_t)row * ldc + col] = c[r];
  }
}

// ---------------- per-(node,head) attention logits --------------------------
__global__ void k_alpha(const float* __restrict__ h,
                        const float* __restrict__ aw_src,
                        const float* __restrict__ aw_dst,
                        float* __restrict__ asrc, float* __restrict__ adst,
                        long long N, int H, int C) {
  long long i = blockIdx.x * 256LL + threadIdx.x;
  if (i >= N * H) return;
  long long n = i / H;
  int hh = (int)(i % H);
  const float* row = h + (size_t)n * H * C + (size_t)hh * C;
  const float* ws  = aw_src + hh * C;
  const float* wd  = aw_dst + hh * C;
  float ss = 0.f, sd = 0.f;
  for (int c = 0; c < C; ++c) { float v = row[c]; ss += v * ws[c]; sd += v * wd[c]; }
  asrc[i] = ss;
  adst[i] = sd;
}

static __device__ __forceinline__ void edge_sd(const int* __restrict__ ei,
                                               long long E, long long e,
                                               int& s, int& d) {
  if (e < E) { s = ei[e]; d = ei[E + e]; }
  else       { s = d = (int)(e - E); }          // self loop
}

// ---------------- edge pass 1: segment max ---------------------------------
__global__ void k_edge_max(const int* __restrict__ ei, long long E, long long N,
                           const float* __restrict__ asrc,
                           const float* __restrict__ adst,
                           unsigned* __restrict__ emax, int H) {
  long long i = blockIdx.x * 256LL + threadIdx.x;
  long long tot = (E + N) * H;
  if (i >= tot) return;
  long long e = i / H; int hh = (int)(i % H);
  int s, d; edge_sd(ei, E, e, s, d);
  float v = asrc[(size_t)s * H + hh] + adst[(size_t)d * H + hh];
  v = (v > 0.f) ? v : NEG_SLOPE * v;
  atomicMax(&emax[(size_t)d * H + hh], f2ord(v));
}

// ---------------- edge pass 2: segment exp-sum -----------------------------
__global__ void k_edge_sum(const int* __restrict__ ei, long long E, long long N,
                           const float* __restrict__ asrc,
                           const float* __restrict__ adst,
                           const unsigned* __restrict__ emax,
                           float* __restrict__ denom, int H) {
  long long i = blockIdx.x * 256LL + threadIdx.x;
  long long tot = (E + N) * H;
  if (i >= tot) return;
  long long e = i / H; int hh = (int)(i % H);
  int s, d; edge_sd(ei, E, e, s, d);
  size_t dH = (size_t)d * H + hh;
  float v = asrc[(size_t)s * H + hh] + adst[dH];
  v = (v > 0.f) ? v : NEG_SLOPE * v;
  atomicAdd(&denom[dH], __expf(v - ord2f(emax[dH])));
}

// ---------------- edge pass 3: weighted scatter-add (heavy) ----------------
// One wave per (edge, head); lanes stripe the C channels. Accumulator is
// ~102 MB -> resident in the 192 MB L2, so f32 atomics stay on-chip.
__global__ void k_edge_agg(const int* __restrict__ ei, long long E, long long N,
                           const float* __restrict__ asrc,
                           const float* __restrict__ adst,
                           const unsigned* __restrict__ emax,
                           const float* __restrict__ denom,
                           const float* __restrict__ h,
                           float* __restrict__ acc, int H, int C) {
  long long w = blockIdx.x * 8LL + (threadIdx.x >> 5);
  int lane = threadIdx.x & 31;
  long long tot = (E + N) * H;
  if (w >= tot) return;
  long long e = w / H; int hh = (int)(w % H);
  int s, d; edge_sd(ei, E, e, s, d);
  size_t dH = (size_t)d * H + hh;
  float v = asrc[(size_t)s * H + hh] + adst[dH];
  v = (v > 0.f) ? v : NEG_SLOPE * v;
  float wgt = __expf(v - ord2f(emax[dH])) / (denom[dH] + 1e-16f);
  const float* hs = h   + (size_t)s * H * C + (size_t)hh * C;
  float*       ad = acc + (size_t)d * H * C + (size_t)hh * C;
  for (int c = lane; c < C; c += 32)
    atomicAdd(&ad[c], hs[c] * wgt);
}

// ------------- finalize (layers 1,2): bias + ELU -> bf16 for next GEMM -----
__global__ void k_finalize_elu_bf16(const float* __restrict__ acc,
                                    const float* __restrict__ b,
                                    __bf16* __restrict__ xbf,
                                    long long N, int F) {
  long long i = blockIdx.x * 256LL + threadIdx.x;
  if (i >= N * F) return;
  int j = (int)(i % F);
  float v = acc[i] + b[j];
  v = (v > 0.f) ? v : (__expf(v) - 1.0f);
  xbf[i] = (__bf16)v;
}

// ------------- finalize (layer 3): bias, mean over 1 head -> logits --------
__global__ void k_finalize_out(const float* __restrict__ acc,
                               const float* __restrict__ b,
                               float* __restrict__ out, long long N, int F) {
  long long i = blockIdx.x * 256LL + threadIdx.x;
  if (i >= N * F) return;
  int j = (int)(i % F);
  out[i] = acc[i] + b[j];
}

// ---------------------------------------------------------------------------
static inline unsigned nblk(long long threads) {
  return (unsigned)((threads + 255) / 256);
}

extern "C" void kernel_launch(void* const* d_in, const int* in_sizes, int n_in,
                              void* d_out, int out_size, void* d_ws, size_t ws_size,
                              hipStream_t stream) {
  const float* x   = (const float*)d_in[0];
  const int*   ei  = (const int*)  d_in[1];
  const float* W1  = (const float*)d_in[2];
  const float* as1 = (const float*)d_in[3];
  const float* ad1 = (const float*)d_in[4];
  const float* b1  = (const float*)d_in[5];
  const float* W2  = (const float*)d_in[6];
  const float* as2 = (const float*)d_in[7];
  const float* ad2 = (const float*)d_in[8];
  const float* b2  = (const float*)d_in[9];
  const float* W3  = (const float*)d_in[10];
  const float* as3 = (const float*)d_in[11];
  const float* ad3 = (const float*)d_in[12];
  const float* b3  = (const float*)d_in[13];
  float* out = (float*)d_out;

  const long long N = in_sizes[0] / 12;   // 50000
  const long long E = in_sizes[1] / 2;    // 600000
  const long long EP = E + N;             // edges incl. self loops
  const int H = 8, C = 64, F = 512;       // heads, per-head dim, hidden

  // -------- workspace carve-up (~263 MB) --------
  char* ws = (char*)d_ws;
  float*    h    = (float*)ws;    ws += (size_t)N * F * 4;   // GEMM output
  float*    acc  = (float*)ws;    ws += (size_t)N * F * 4;   // aggregation
  __bf16*   xbf  = (__bf16*)ws;   ws += (size_t)N * F * 2;   // bf16 activations
  __bf16*   wt   = (__bf16*)ws;   ws += (size_t)512 * 512 * 2; // bf16 weights^T
  float*    asrc = (float*)ws;    ws += (size_t)N * H * 4;
  float*    adst = (float*)ws;    ws += (size_t)N * H * 4;
  unsigned* emax = (unsigned*)ws; ws += (size_t)N * H * 4;
  float*    den  = (float*)ws;    ws += (size_t)N * H * 4;

  const int Mt = (int)(N / 16);           // 50000 = 16 * 3125 exactly

  // =========================== layer 1: 12 -> 512 ==========================
  k_cvt_x <<<nblk(N * 32), 256, 0, stream>>>(x, xbf, N);
  k_cvt_wt<<<nblk(512 * 32), 256, 0, stream>>>(W1, wt, 12, 512, 32, 512);
  {
    long long waves = (long long)Mt * 32;
    k_gemm_bf16<<<nblk(waves * 32), 256, 0, stream>>>(
        xbf, wt, h, Mt, 32, /*Ksteps=*/1, /*lda=*/32, /*ldb=*/32, /*ldc=*/F, F);
  }
  k_alpha<<<nblk(N * H), 256, 0, stream>>>(h, as1, ad1, asrc, adst, N, H, C);
  k_zero32<<<nblk(N * H), 256, 0, stream>>>(emax, N * H);          // 0 == -inf key
  k_zero32<<<nblk(N * H), 256, 0, stream>>>((unsigned*)den, N * H);
  k_zero32<<<nblk(N * F), 256, 0, stream>>>((unsigned*)acc, N * F);
  k_edge_max<<<nblk(EP * H), 256, 0, stream>>>(ei, E, N, asrc, adst, emax, H);
  k_edge_sum<<<nblk(EP * H), 256, 0, stream>>>(ei, E, N, asrc, adst, emax, den, H);
  k_edge_agg<<<nblk(EP * H * 32), 256, 0, stream>>>(ei, E, N, asrc, adst, emax,
                                                    den, h, acc, H, C);
  k_finalize_elu_bf16<<<nblk(N * F), 256, 0, stream>>>(acc, b1, xbf, N, F);

  // =========================== layer 2: 512 -> 512 =========================
  k_cvt_wt<<<nblk(512 * 512), 256, 0, stream>>>(W2, wt, 512, 512, 512, 512);
  {
    long long waves = (long long)Mt * 32;
    k_gemm_bf16<<<nblk(waves * 32), 256, 0, stream>>>(
        xbf, wt, h, Mt, 32, /*Ksteps=*/16, /*lda=*/512, /*ldb=*/512, F, F);
  }
  k_alpha<<<nblk(N * H), 256, 0, stream>>>(h, as2, ad2, asrc, adst, N, H, C);
  k_zero32<<<nblk(N * H), 256, 0, stream>>>(emax, N * H);
  k_zero32<<<nblk(N * H), 256, 0, stream>>>((unsigned*)den, N * H);
  k_zero32<<<nblk(N * F), 256, 0, stream>>>((unsigned*)acc, N * F);
  k_edge_max<<<nblk(EP * H), 256, 0, stream>>>(ei, E, N, asrc, adst, emax, H);
  k_edge_sum<<<nblk(EP * H), 256, 0, stream>>>(ei, E, N, asrc, adst, emax, den, H);
  k_edge_agg<<<nblk(EP * H * 32), 256, 0, stream>>>(ei, E, N, asrc, adst, emax,
                                                    den, h, acc, H, C);
  k_finalize_elu_bf16<<<nblk(N * F), 256, 0, stream>>>(acc, b2, xbf, N, F);

  // ==================== layer 3: 512 -> 3, single head =====================
  k_cvt_wt<<<nblk(16 * 512), 256, 0, stream>>>(W3, wt, 512, 3, 512, 16);
  {
    long long waves = (long long)Mt * 1;   // one 16-wide tile, keep 3 cols
    k_gemm_bf16<<<nblk(waves * 32), 256, 0, stream>>>(
        xbf, wt, h, Mt, 1, /*Ksteps=*/16, /*lda=*/512, /*ldb=*/512,
        /*ldc=*/3, /*nlim=*/3);
  }
  k_alpha<<<nblk(N * 1), 256, 0, stream>>>(h, as3, ad3, asrc, adst, N, 1, 3);
  k_zero32<<<nblk(N * 1), 256, 0, stream>>>(emax, N);
  k_zero32<<<nblk(N * 1), 256, 0, stream>>>((unsigned*)den, N);
  k_zero32<<<nblk(N * 3), 256, 0, stream>>>((unsigned*)acc, N * 3);
  k_edge_max<<<nblk(EP), 256, 0, stream>>>(ei, E, N, asrc, adst, emax, 1);
  k_edge_sum<<<nblk(EP), 256, 0, stream>>>(ei, E, N, asrc, adst, emax, den, 1);
  k_edge_agg<<<nblk(EP * 32), 256, 0, stream>>>(ei, E, N, asrc, adst, emax,
                                                den, h, acc, 1, 3);
  k_finalize_out<<<nblk(N * 3), 256, 0, stream>>>(acc, b3, out, N, 3);
}
